// CustomRNN_27711128994073
// MI455X (gfx1250) — matrix operand order repaired and checked by reading, not compiled, not run on hardware
//
#include <hip/hip_runtime.h>
#include <hip/hip_bf16.h>

// ---------------------------------------------------------------------------
// CustomRNN on MI455X (gfx1250, wave32, WMMA).
//   Phase 0: transpose+convert W1,W2 -> bf16 (WMMA B-operand friendly layout)
//   Phase 1: h = tanh(x @ W1 + b1) via v_wmma_f32_16x16x32_bf16, h -> d_out
//   Phase 2: sequential scan y_t = h_t + tanh(state @ W2 + b2), in-place on d_out
//            W2 fragments pinned in VGPRs across all 512 steps.
// ---------------------------------------------------------------------------

typedef __attribute__((ext_vector_type(16))) __bf16 bf16x16;
typedef __attribute__((ext_vector_type(8)))  __bf16 bf16x8;
typedef __attribute__((ext_vector_type(8)))  float  f32x8;

#define B_DIM 256
#define T_DIM 512
#define D_DIM 256
#define U_DIM 256

#define BM    32                  // M rows per workgroup in projection
#define LDK   (D_DIM + 8)         // padded LDS row stride (bf16 elems)
#define LDU   (U_DIM + 8)         // padded LDS row stride for state (bf16 elems)

// Hardware tanh if this toolchain exposes it for gfx1250 (V_TANH_F32, TRANS op);
// falls back to libm otherwise.
#if __has_builtin(__builtin_amdgcn_tanhf)
__device__ __forceinline__ float fast_tanh(float x) { return __builtin_amdgcn_tanhf(x); }
#elif __has_builtin(__builtin_amdgcn_tanh_f32)
__device__ __forceinline__ float fast_tanh(float x) { return __builtin_amdgcn_tanh_f32(x); }
#else
__device__ __forceinline__ float fast_tanh(float x) { return tanhf(x); }
#endif

// A fragment (16x32 bf16): lane<16 -> K=[k0..k0+15] contiguous; lane>=16 -> [k0+16..k0+31]
// caller passes p = base + k0 + hi*16; 16 contiguous bf16.
__device__ __forceinline__ bf16x16 ld_frag_a(const __bf16* p) {
    bf16x8 lo = *(const bf16x8*)(p);
    bf16x8 hi = *(const bf16x8*)(p + 8);
    return __builtin_shufflevector(lo, hi, 0,1,2,3,4,5,6,7,8,9,10,11,12,13,14,15);
}

// B fragment (32x16 bf16): lane<16 -> K=[k0..k0+15] contiguous; lane>=16 -> [k0+16..k0+31]
// caller passes p = WT_row(c) + k0 + hi*16; 16 contiguous bf16.
__device__ __forceinline__ bf16x16 ld_frag_b(const __bf16* p) {
    bf16x8 lo = *(const bf16x8*)(p);
    bf16x8 hi = *(const bf16x8*)(p + 8);
    return __builtin_shufflevector(lo, hi, 0,1,2,3,4,5,6,7,8,9,10,11,12,13,14,15);
}

__device__ __forceinline__ f32x8 wmma_bf16(bf16x16 a, bf16x16 b, f32x8 c) {
    return __builtin_amdgcn_wmma_f32_16x16x32_bf16(
        /*neg_a=*/false, a, /*neg_b=*/false, b,
        /*c_mod=*/(short)0, c, /*reuse_a=*/false, /*reuse_b=*/false);
}

// ---------------------------------------------------------------------------
// Phase 0: W1T[u][d] = bf16(W1[d][u]); W2T[n][k] = bf16(W2[k][n])
// ---------------------------------------------------------------------------
__global__ __launch_bounds__(256) void rnn_prep_weights(
        const float* __restrict__ W1, const float* __restrict__ W2,
        __bf16* __restrict__ W1T, __bf16* __restrict__ W2T) {
    int i = blockIdx.x * 256 + threadIdx.x;   // output index u*256 + d
    int u = i >> 8;
    int d = i & 255;
    W1T[i] = (__bf16)W1[d * U_DIM + u];
    W2T[i] = (__bf16)W2[d * U_DIM + u];
}

// ---------------------------------------------------------------------------
// Phase 1: h[m][u] = tanh(sum_k x[m][k]*W1[k][u] + b1[u]),  m in [0, B*T)
// One workgroup = 8 waves = 32 rows x 256 cols; x tile staged in LDS as bf16.
// ---------------------------------------------------------------------------
__global__ __launch_bounds__(256) void rnn_proj(
        const float* __restrict__ x, const __bf16* __restrict__ W1T,
        const float* __restrict__ b1, float* __restrict__ h /* == d_out */) {
    __shared__ __bf16 sA[BM * LDK];

    const int  tid = threadIdx.x;
    const long m0  = (long)blockIdx.x * BM;

    // Cooperative load: 32 rows x 256 f32 -> bf16 LDS. Thread t: row t/8, 32 cols.
    {
        int r  = tid >> 3;
        int kb = (tid & 7) * 32;
        const f32x8* src = (const f32x8*)(x + (m0 + r) * D_DIM + kb);
        bf16x8*      dst = (bf16x8*)(sA + r * LDK + kb);
#pragma unroll
        for (int i = 0; i < 4; ++i)
            dst[i] = __builtin_convertvector(src[i], bf16x8);
    }
    __syncthreads();

    const int wave = tid >> 5;
    const int lane = tid & 31;
    const int r    = lane & 15;
    const int hi   = lane >> 4;
    const int n0   = wave * 32;

    f32x8 acc[2][2] = {};   // [m-tile][n-tile]

#pragma unroll
    for (int k0 = 0; k0 < D_DIM; k0 += 32) {
        bf16x16 a[2];
#pragma unroll
        for (int mt = 0; mt < 2; ++mt)
            a[mt] = ld_frag_a(sA + (mt * 16 + r) * LDK + k0 + hi * 16);
#pragma unroll
        for (int nt = 0; nt < 2; ++nt) {
            bf16x16 b = ld_frag_b(W1T + (n0 + nt * 16 + r) * D_DIM + k0 + hi * 16);
            acc[0][nt] = wmma_bf16(a[0], b, acc[0][nt]);
            acc[1][nt] = wmma_bf16(a[1], b, acc[1][nt]);
        }
    }

    // Epilogue: bias + tanh, scatter-store h (C layout: row = v + 8*hi, col = r)
#pragma unroll
    for (int nt = 0; nt < 2; ++nt) {
        int   c    = n0 + nt * 16 + r;
        float bias = b1[c];
#pragma unroll
        for (int mt = 0; mt < 2; ++mt) {
#pragma unroll
            for (int v = 0; v < 8; ++v) {
                long row = m0 + mt * 16 + v + 8 * hi;
                h[row * U_DIM + c] = fast_tanh(acc[mt][nt][v] + bias);
            }
        }
    }
}

// ---------------------------------------------------------------------------
// Phase 2: per 16-row batch block, scan over T:
//   y_t = h_t + tanh(state @ W2 + b2); state = y_t
// state double-buffered in LDS (bf16); all W2T fragments held in VGPRs.
// ---------------------------------------------------------------------------
__global__ __launch_bounds__(256) void rnn_scan(
        const __bf16* __restrict__ W2T, const float* __restrict__ b2,
        float* __restrict__ y /* == d_out, holds h on entry */) {
    __shared__ __bf16 sS[2][16 * LDU];

    const int tid  = threadIdx.x;
    const int b0   = blockIdx.x * 16;   // batch rows [b0, b0+16)
    const int wave = tid >> 5;
    const int lane = tid & 31;
    const int r    = lane & 15;
    const int hi   = lane >> 4;
    const int n0   = wave * 32;

    // Zero initial state buffer 0.
    for (int i = tid; i < 16 * LDU; i += 256) sS[0][i] = (__bf16)0.0f;

    // Pin all W2T fragments this wave needs in VGPRs (invariant over t):
    // 8 k-steps x 2 n-tiles x 32B = 128 VGPRs.
    bf16x16 bw[8][2];
#pragma unroll
    for (int kk = 0; kk < 8; ++kk)
#pragma unroll
        for (int nt = 0; nt < 2; ++nt)
            bw[kk][nt] = ld_frag_b(W2T + (n0 + nt * 16 + r) * U_DIM + kk * 32 + hi * 16);

    const float bias[2] = { b2[n0 + r], b2[n0 + 16 + r] };

    // Per-row output pointers (col fixed per lane); bumped by U each step.
    float* yp[2][8];
#pragma unroll
    for (int nt = 0; nt < 2; ++nt)
#pragma unroll
        for (int v = 0; v < 8; ++v) {
            int row = v + 8 * hi;
            yp[nt][v] = y + (long)(b0 + row) * T_DIM * U_DIM + (n0 + nt * 16 + r);
        }

    __syncthreads();

    for (int t = 0; t < T_DIM; ++t) {
        const __bf16* S = sS[t & 1];
        f32x8 acc[2] = {};
#pragma unroll
        for (int kk = 0; kk < 8; ++kk) {
            bf16x16 a = ld_frag_a(S + r * LDU + kk * 32 + hi * 16);
            acc[0] = wmma_bf16(a, bw[kk][0], acc[0]);
            acc[1] = wmma_bf16(a, bw[kk][1], acc[1]);
        }

        __bf16* Snext = sS[(t + 1) & 1];
#pragma unroll
        for (int nt = 0; nt < 2; ++nt) {
            int c = n0 + nt * 16 + r;
#pragma unroll
            for (int v = 0; v < 8; ++v) {
                int   row = v + 8 * hi;
                float yv  = *yp[nt][v] + fast_tanh(acc[nt][v] + bias[nt]);
                *yp[nt][v] = yv;
                Snext[row * LDU + c] = (__bf16)yv;
                if (t + 1 < T_DIM)   // prefetch next step's h (global_prefetch_b8)
                    __builtin_prefetch(yp[nt][v] + U_DIM, 0, 0);
                yp[nt][v] += U_DIM;
            }
        }
        __syncthreads();   // Snext complete; old S no longer read
    }
}

// ---------------------------------------------------------------------------
extern "C" void kernel_launch(void* const* d_in, const int* in_sizes, int n_in,
                              void* d_out, int out_size, void* d_ws, size_t ws_size,
                              hipStream_t stream) {
    const float* x  = (const float*)d_in[0];
    const float* W1 = (const float*)d_in[1];
    const float* b1 = (const float*)d_in[2];
    const float* W2 = (const float*)d_in[3];
    const float* b2 = (const float*)d_in[4];
    float* out = (float*)d_out;

    __bf16* W1T = (__bf16*)d_ws;                 // 256*256 bf16 = 128 KB
    __bf16* W2T = W1T + D_DIM * U_DIM;           // next 128 KB

    rnn_prep_weights<<<(D_DIM * U_DIM) / 256, 256, 0, stream>>>(W1, W2, W1T, W2T);
    rnn_proj<<<(B_DIM * T_DIM) / BM, 256, 0, stream>>>(x, W1T, b1, out);
    rnn_scan<<<B_DIM / 16, 256, 0, stream>>>(W2T, b2, out);
}
